// EnglishOOVPredictor_84413287236027
// MI455X (gfx1250) — compile-verified, matrix operand order checked by reading, not run on hardware
//
#include <hip/hip_runtime.h>
#include <hip/hip_bf16.h>
#include <cstdint>

// ---------------------------------------------------------------------------
// GRU seq2seq on gfx1250: bf16 WMMA (v_wmma_f32_16x16x32_bf16) for all GEMMs,
// TDM (tensor_load_to_lds) staging of the recurrent weight tile into LDS with
// bank-conflict padding, fused gate epilogue, fp32 accumulate.
// wave32, 8 waves / block.
// ---------------------------------------------------------------------------

typedef __bf16 bfv8   __attribute__((ext_vector_type(8)));
typedef __bf16 bfv16  __attribute__((ext_vector_type(16)));
typedef float  fv8    __attribute__((ext_vector_type(8)));
typedef unsigned int uv4 __attribute__((ext_vector_type(4)));
typedef int    iv4    __attribute__((ext_vector_type(4)));
typedef int    iv8    __attribute__((ext_vector_type(8)));

#define DEVI static __device__ __forceinline__

#if __has_include(<hip/amd_detail/amd_gfx1250_TDM.h>)
#define TDM_6ARG 1
#endif
#if __has_builtin(__builtin_amdgcn_tensor_load_to_lds)
#define HAVE_TDM 1
#endif

// Load one 16x32 bf16 A/B fragment slice for this lane from a row-major
// [rows, ldk] matrix (global or LDS). Caller passes p = row_ptr + 32*kk +
// 8*(lane>>4). VGPR0..3 <- K = k0..k0+7 (16B), VGPR4..7 <- K = k0+16..+23.
DEVI bfv16 load_frag(const __bf16* __restrict__ p) {
  bfv8 lo = *(const bfv8*)(p);
  bfv8 hi = *(const bfv8*)(p + 16);
  return __builtin_shufflevector(lo, hi, 0,1,2,3,4,5,6,7,8,9,10,11,12,13,14,15);
}

DEVI fv8 wmma_bf16(bfv16 a, bfv16 b, fv8 c) {
  return __builtin_amdgcn_wmma_f32_16x16x32_bf16(
      /*neg_a=*/false, a, /*neg_b=*/false, b,
      /*c_mod=*/(short)0, c, /*reuse_a=*/false, /*reuse_b=*/false);
}

DEVI float sigmoid_f(float x) { return 1.0f / (1.0f + __expf(-x)); }

// ---------------------------------------------------------------------------
// TDM: load a 3D weight tile  [z=3 gates][y=16 rows][x=K elems] (bf16) from
// global (row stride K, gate stride H*K) into LDS at lds_byte_off, padding
// 4 DWORDs after every 256 DWORDs (1024B row -> 1040B LDS pitch).
// D# layout per CDNA5 ISA.
// ---------------------------------------------------------------------------
#if HAVE_TDM
DEVI void tdm_load_w3(const __bf16* gsrc, unsigned lds_byte_off,
                      int K, int gate_stride_elems) {
  unsigned long long ga = (unsigned long long)(uintptr_t)gsrc;
  uv4 g0;
  g0.x = 1u;                                            // count=1 (valid)
  g0.y = lds_byte_off;                                  // lds_addr
  g0.z = (unsigned)(ga & 0xFFFFFFFFu);                  // global_addr[31:0]
  g0.w = (unsigned)((ga >> 32) & 0x1FFFFFFu)            // global_addr[56:32]
       | (2u << 30);                                    // type=2 ("image")
  iv8 g1;
  g1[0] = (1 << 16)                                     // data_size = 2B
        | (1 << 20)                                     // pad_enable
        | (7 << 22)                                     // pad_interval: 256 DW
        | (3 << 25);                                    // pad_amount: 4 DW
  g1[1] = (int)((unsigned)(K & 0xFFFF) << 16);          // tensor_dim0 lo16
  g1[2] = (int)((K >> 16) & 0xFFFF) | (16 << 16);       // dim0 hi | tensor_dim1=16
  g1[3] = (int)((unsigned)(K & 0xFFFF) << 16);          // tile_dim0 = K
  g1[4] = 16 | (3 << 16);                               // tile_dim1=16, tile_dim2=3
  g1[5] = K;                                            // tensor_dim0_stride lo32
  g1[6] = (int)(((unsigned)gate_stride_elems & 0xFFFFu) << 16); // dim1_stride lo16
  g1[7] = (int)((unsigned)gate_stride_elems >> 16);     // dim1_stride hi32
  iv4 g2; g2[0] = 3; g2[1] = 1; g2[2] = 0; g2[3] = 0;   // tensor_dim2=3
  iv4 g3 = {0, 0, 0, 0};
#if TDM_6ARG
  iv8 gz = {0, 0, 0, 0, 0, 0, 0, 0};
  __builtin_amdgcn_tensor_load_to_lds(g0, g1, g2, g3, gz, 0);
#else
  __builtin_amdgcn_tensor_load_to_lds(g0, g1, g2, g3, 0);
#endif
}
#endif

// LDS pitch for one 512-elem bf16 row after TDM padding: 1024 + 16 B.
constexpr int WHH_PITCH = 520;            // elements (1040 bytes)
constexpr int WHH_PLANE = 16 * WHH_PITCH; // one gate plane
constexpr int WHH_ELEMS = 3 * WHH_PLANE;  // 49,920 B total

// ---------------------------------------------------------------------------
// Fused GRU step: h_new = GRU(emb[ids], h). One wave = 16(M) x 16(N in H)
// output tile, 4 accumulators (r, z, n_ih, n_hh). Block = 8 waves -> M=128.
// Grid = (H/16, B/128). Optionally captures h at step == clip(len-1,0,31).
// ---------------------------------------------------------------------------
__launch_bounds__(256)
__global__ void gru_step_kernel(
    const int*   __restrict__ ids, int ids_stride,    // token id per batch row
    const __bf16* __restrict__ emb,                   // [V, E] bf16
    const __bf16* __restrict__ w_ih,                  // [3H, E] bf16
    const __bf16* __restrict__ w_hh,                  // [3H, H] bf16
    const float* __restrict__ b_ih,                   // [3H]
    const float* __restrict__ b_hh,                   // [3H]
    const float* __restrict__ h_in_f,                 // [B, H] f32
    const __bf16* __restrict__ h_in_b,                // [B, H] bf16
    float*  __restrict__ h_out_f,
    __bf16* __restrict__ h_out_b,
    const int* __restrict__ seq_len,                  // null => no capture
    int step,
    float*  __restrict__ h_cap_f,
    __bf16* __restrict__ h_cap_b)
{
  constexpr int E = 256, H = 512;
  __shared__ __align__(16) __bf16 swhh[WHH_ELEMS];   // [3][16][520] bf16

  const int lane = threadIdx.x & 31;
  const int wave = threadIdx.x >> 5;
  const int half = lane >> 4;
  const int l16  = lane & 15;
  const int m0   = blockIdx.y * 128 + wave * 16;   // batch-row tile base
  const int n0   = blockIdx.x * 16;                // column tile base (H space)
  const int rowA = m0 + l16;
  const int kA   = 8 * half;

  // ---- stage w_hh tile [3 gates][16 rows n0..n0+15][K=H] into LDS ----
#if HAVE_TDM
  if (wave == 0) {
    tdm_load_w3(w_hh + (size_t)n0 * H,
                (unsigned)(uintptr_t)(void*)swhh,   // flat LDS addr[31:0]
                H, H * H /* gate z-stride in elems */);
    __builtin_amdgcn_s_wait_tensorcnt(0);
  }
#else
  // cooperative fallback: 48 rows x 64 16B-chunks, padded pitch 1040B
  for (int idx = threadIdx.x; idx < 3 * 16 * 64; idx += 256) {
    const int row = idx >> 6, chunk = idx & 63;
    const int gate = row >> 4, r = row & 15;
    const bfv8* src =
        (const bfv8*)(w_hh + ((size_t)gate * H + n0 + r) * H) + chunk;
    *(bfv8*)((char*)swhh + row * (WHH_PITCH * 2) + chunk * 16) = *src;
  }
#endif
  __syncthreads();

  const int tok = ids[rowA * ids_stride];
  const __bf16* xrow = emb + (size_t)tok * E;
  const __bf16* hrow = h_in_b + (size_t)rowA * H;

  fv8 acc_r = {}, acc_z = {}, acc_ni = {}, acc_nh = {};

  // ---- x @ W_ih^T, K = E = 256 (8 WMMA k-chunks x 3 gates), B from global ----
  const __bf16* wr = w_ih + (size_t)(0 * H + n0 + l16) * E;
  const __bf16* wz = w_ih + (size_t)(1 * H + n0 + l16) * E;
  const __bf16* wn = w_ih + (size_t)(2 * H + n0 + l16) * E;
#pragma unroll
  for (int kk = 0; kk < E / 32; ++kk) {
    const int k0 = kk * 32 + kA;
    bfv16 a = load_frag(xrow + k0);
    acc_r  = wmma_bf16(a, load_frag(wr + k0), acc_r);
    acc_z  = wmma_bf16(a, load_frag(wz + k0), acc_z);
    acc_ni = wmma_bf16(a, load_frag(wn + k0), acc_ni);
  }

  // ---- h @ W_hh^T, K = H = 512 (16 WMMA k-chunks x 3 gates), B from LDS ----
  const __bf16* vr = swhh + 0 * WHH_PLANE + l16 * WHH_PITCH;
  const __bf16* vz = swhh + 1 * WHH_PLANE + l16 * WHH_PITCH;
  const __bf16* vn = swhh + 2 * WHH_PLANE + l16 * WHH_PITCH;
#pragma unroll
  for (int kk = 0; kk < H / 32; ++kk) {
    const int k0 = kk * 32 + kA;
    __builtin_prefetch(hrow + k0 + 64, 0, 1);   // global_prefetch_b8
    bfv16 a = load_frag(hrow + k0);
    acc_r  = wmma_bf16(a, load_frag(vr + k0), acc_r);
    acc_z  = wmma_bf16(a, load_frag(vz + k0), acc_z);
    acc_nh = wmma_bf16(a, load_frag(vn + k0), acc_nh);
  }

  // ---- gate epilogue. C/D layout: element i -> (m = 8*half+i, n = l16) ----
  const int gn   = n0 + l16;
  const float br  = b_ih[gn]         + b_hh[gn];
  const float bz  = b_ih[H + gn]     + b_hh[H + gn];
  const float bni = b_ih[2 * H + gn];
  const float bnh = b_hh[2 * H + gn];
#pragma unroll
  for (int i = 0; i < 8; ++i) {
    const int   gm    = m0 + 8 * half + i;
    const float r     = sigmoid_f(acc_r[i] + br);
    const float z     = sigmoid_f(acc_z[i] + bz);
    const float nn    = tanhf(acc_ni[i] + bni + r * (acc_nh[i] + bnh));
    const float hprev = h_in_f[(size_t)gm * H + gn];
    const float hnew  = (1.0f - z) * nn + z * hprev;
    h_out_f[(size_t)gm * H + gn] = hnew;
    h_out_b[(size_t)gm * H + gn] = (__bf16)hnew;
    if (seq_len) {
      int li = seq_len[gm] - 1;
      li = li < 0 ? 0 : (li > 31 ? 31 : li);
      if (li == step) {
        h_cap_f[(size_t)gm * H + gn] = hnew;
        h_cap_b[(size_t)gm * H + gn] = (__bf16)hnew;
      }
    }
  }
}

// ---------------------------------------------------------------------------
// logits = h @ fc_w^T + fc_b, then first-index argmax + EOS masking.
// Block: M=16 rows x full V=128 cols (8 waves x N=16, 16 WMMAs each).
// ---------------------------------------------------------------------------
__launch_bounds__(256)
__global__ void logits_argmax_kernel(
    const __bf16* __restrict__ h_b,   // [B, H] bf16
    const __bf16* __restrict__ fc_w,  // [V, H] bf16
    const float*  __restrict__ fc_b,  // [V]
    int* __restrict__ finished,       // [B]
    int* __restrict__ pred,           // [B]
    int* __restrict__ out,            // [B, T] int32
    int step)
{
  constexpr int H = 512, V = 128, T = 32, EOS = 3;
  __shared__ float lds[16 * V];

  const int lane = threadIdx.x & 31;
  const int wave = threadIdx.x >> 5;
  const int half = lane >> 4;
  const int l16  = lane & 15;
  const int m0   = blockIdx.x * 16;
  const int n0   = wave * 16;
  const int kA   = 8 * half;

  const __bf16* hrow = h_b  + (size_t)(m0 + l16) * H;
  const __bf16* wrow = fc_w + (size_t)(n0 + l16) * H;

  fv8 acc = {};
#pragma unroll
  for (int kk = 0; kk < H / 32; ++kk) {
    const int k0 = kk * 32 + kA;
    acc = wmma_bf16(load_frag(hrow + k0), load_frag(wrow + k0), acc);
  }

  const int   gn   = n0 + l16;
  const float bias = fc_b[gn];
#pragma unroll
  for (int i = 0; i < 8; ++i)
    lds[(8 * half + i) * V + gn] = acc[i] + bias;
  __syncthreads();

  if (threadIdx.x < 16) {
    const int b = m0 + threadIdx.x;
    const float* row = lds + threadIdx.x * V;
    int best = 0; float bv = row[0];
    for (int j = 1; j < V; ++j) {
      float v = row[j];
      if (v > bv) { bv = v; best = j; }   // first max (jnp.argmax semantics)
    }
    int fin = finished[b];
    int p   = fin ? EOS : best;
    finished[b] = fin | (p == EOS);
    pred[b]     = p;
    out[(size_t)b * T + step] = p;
  }
}

// ---------------------------------------------------------------------------
// Small utility kernels
// ---------------------------------------------------------------------------
__global__ void cvt_bf16_kernel(const float* __restrict__ s, __bf16* __restrict__ d, int n) {
  int i = blockIdx.x * blockDim.x + threadIdx.x;
  if (i < n) d[i] = (__bf16)s[i];
}
__global__ void fill_f32_kernel(float* p, float v, int n) {
  int i = blockIdx.x * blockDim.x + threadIdx.x;
  if (i < n) p[i] = v;
}
__global__ void fill_bf16_kernel(__bf16* p, float v, int n) {
  int i = blockIdx.x * blockDim.x + threadIdx.x;
  if (i < n) p[i] = (__bf16)v;
}
__global__ void fill_i32_kernel(int* p, int v, int n) {
  int i = blockIdx.x * blockDim.x + threadIdx.x;
  if (i < n) p[i] = v;
}

// ---------------------------------------------------------------------------
// Host orchestration (graph-capture safe: only kernel launches on `stream`)
// ---------------------------------------------------------------------------
extern "C" void kernel_launch(void* const* d_in, const int* in_sizes, int n_in,
                              void* d_out, int out_size, void* d_ws, size_t ws_size,
                              hipStream_t stream) {
  constexpr int B = 8192, S = 32, E = 256, H = 512, T = 32;
  constexpr int ENC_V = 100, DEC_V = 128, G = 3 * H;

  const int*   input_ids = (const int*)d_in[0];
  const int*   input_len = (const int*)d_in[1];
  const float* enc_emb   = (const float*)d_in[2];
  const float* enc_w_ih  = (const float*)d_in[3];
  const float* enc_w_hh  = (const float*)d_in[4];
  const float* enc_b_ih  = (const float*)d_in[5];
  const float* enc_b_hh  = (const float*)d_in[6];
  const float* dec_emb   = (const float*)d_in[7];
  const float* dec_w_ih  = (const float*)d_in[8];
  const float* dec_w_hh  = (const float*)d_in[9];
  const float* dec_b_ih  = (const float*)d_in[10];
  const float* dec_b_hh  = (const float*)d_in[11];
  const float* fc_w      = (const float*)d_in[12];
  const float* fc_b      = (const float*)d_in[13];
  int* out = (int*)d_out;   // reference output dtype is int32 [B, 32]

  // --- carve workspace (everything rewritten every call) ---
  char* w = (char*)d_ws;
  auto carve = [&](size_t bytes) {
    char* p = w;
    w += (bytes + 255) & ~size_t(255);
    return (void*)p;
  };
  float*  hA_f = (float*) carve((size_t)B * H * 4);
  float*  hB_f = (float*) carve((size_t)B * H * 4);
  float*  hC_f = (float*) carve((size_t)B * H * 4);   // encoder capture
  __bf16* hA_b = (__bf16*)carve((size_t)B * H * 2);
  __bf16* hB_b = (__bf16*)carve((size_t)B * H * 2);
  __bf16* hC_b = (__bf16*)carve((size_t)B * H * 2);
  __bf16* wEih = (__bf16*)carve((size_t)G * E * 2);
  __bf16* wEhh = (__bf16*)carve((size_t)G * H * 2);
  __bf16* wDih = (__bf16*)carve((size_t)G * E * 2);
  __bf16* wDhh = (__bf16*)carve((size_t)G * H * 2);
  __bf16* wFC  = (__bf16*)carve((size_t)DEC_V * H * 2);
  __bf16* embE = (__bf16*)carve((size_t)ENC_V * E * 2);
  __bf16* embD = (__bf16*)carve((size_t)DEC_V * E * 2);
  int*    pred = (int*)   carve((size_t)B * 4);
  int*    fin  = (int*)   carve((size_t)B * 4);

  auto blocks = [](int n) { return (n + 255) / 256; };

  // --- convert weights / embeddings to bf16 ---
  cvt_bf16_kernel<<<blocks(G * E), 256, 0, stream>>>(enc_w_ih, wEih, G * E);
  cvt_bf16_kernel<<<blocks(G * H), 256, 0, stream>>>(enc_w_hh, wEhh, G * H);
  cvt_bf16_kernel<<<blocks(G * E), 256, 0, stream>>>(dec_w_ih, wDih, G * E);
  cvt_bf16_kernel<<<blocks(G * H), 256, 0, stream>>>(dec_w_hh, wDhh, G * H);
  cvt_bf16_kernel<<<blocks(DEC_V * H), 256, 0, stream>>>(fc_w, wFC, DEC_V * H);
  cvt_bf16_kernel<<<blocks(ENC_V * E), 256, 0, stream>>>(enc_emb, embE, ENC_V * E);
  cvt_bf16_kernel<<<blocks(DEC_V * E), 256, 0, stream>>>(dec_emb, embD, DEC_V * E);

  // --- init state: h0 = 0, pred = BOS(2), finished = 0 ---
  fill_f32_kernel <<<blocks(B * H), 256, 0, stream>>>(hA_f, 0.0f, B * H);
  fill_bf16_kernel<<<blocks(B * H), 256, 0, stream>>>(hA_b, 0.0f, B * H);
  fill_i32_kernel <<<blocks(B), 256, 0, stream>>>(pred, 2, B);
  fill_i32_kernel <<<blocks(B), 256, 0, stream>>>(fin, 0, B);

  const dim3 gg(H / 16, B / 128);   // 32 x 64 blocks, 256 threads (8 waves)

  // --- encoder: 32 sequential GRU steps, capture h at last_index ---
  float* hin_f = hA_f;  __bf16* hin_b = hA_b;
  float* hout_f = hB_f; __bf16* hout_b = hB_b;
  for (int t = 0; t < S; ++t) {
    gru_step_kernel<<<gg, 256, 0, stream>>>(
        input_ids + t, S, embE, wEih, wEhh, enc_b_ih, enc_b_hh,
        hin_f, hin_b, hout_f, hout_b,
        input_len, t, hC_f, hC_b);
    float* tf = hin_f; hin_f = hout_f; hout_f = tf;
    __bf16* tb = hin_b; hin_b = hout_b; hout_b = tb;
  }

  // --- decoder: 32 steps of (GRU step -> logits/argmax/EOS) ---
  hin_f = hC_f; hin_b = hC_b;          // start from captured encoder state
  hout_f = hA_f; hout_b = hA_b;        // hA free to reuse
  for (int t = 0; t < T; ++t) {
    gru_step_kernel<<<gg, 256, 0, stream>>>(
        pred, 1, embD, wDih, wDhh, dec_b_ih, dec_b_hh,
        hin_f, hin_b, hout_f, hout_b,
        nullptr, 0, nullptr, nullptr);
    logits_argmax_kernel<<<B / 16, 256, 0, stream>>>(
        hout_b, wFC, fc_b, fin, pred, out, t);
    // ping-pong: next in = this out; next out = hA/hB alternate
    hin_f = hout_f; hin_b = hout_b;
    if (hout_f == hA_f) { hout_f = hB_f; hout_b = hB_b; }
    else                { hout_f = hA_f; hout_b = hA_b; }
  }
}
